// GAT_66194035966463
// MI455X (gfx1250) — matrix operand order repaired and checked by reading, not compiled, not run on hardware
//
#include <hip/hip_runtime.h>

// GATv2 x3 + LayerNorms + final linear for MI455X (gfx1250, wave32).
// Dense projections run on WMMA (bf16 in, fp32 accumulate); edge softmax/
// aggregation is the HBM-bound phase (float4 gathers + global float atomics).

#define N_NODES 50000
#define N_EDGES 800000

typedef __attribute__((ext_vector_type(16))) __bf16 v16bf;
typedef __attribute__((ext_vector_type(8)))  float  v8f;

union AFragU { v16bf v; __bf16 e[16]; };
union BFragU { v16bf v; uint4 q[2]; };
union BF2U   { __bf16 e[2]; unsigned u; };

// ---------------------------------------------------------------------------
// Y[Nrows,M] = bf16(X[Nrows,K]) @ bf16(W[K,M]) + bias, fp32 accumulate.
// Block = 256 threads = 8 waves. Block tile = 128 rows x 64 cols;
// wave w owns a 16x64 strip -> 4 accumulators, 4 WMMAs per K-step from one
// A fragment. K % 32 == 0, M % 64 == 0 for all shapes in this model.
// ---------------------------------------------------------------------------
__global__ void __launch_bounds__(256)
wmma_gemm_bias(const float* __restrict__ X, const float* __restrict__ W,
               const float* __restrict__ bias, float* __restrict__ Y,
               int Nrows, int K, int M) {
  __shared__ alignas(16) __bf16 Bs[64 * 32];            // B tile, [n][k] (transposed)
  const int lane = threadIdx.x & 31;
  const int wave = threadIdx.x >> 5;
  const int half = lane >> 4;                           // 0: lanes 0-15, 1: lanes 16-31
  const int m0 = (blockIdx.x * 8 + wave) * 16;
  const int n0 = blockIdx.y * 64;

  v8f acc0 = {0.f,0.f,0.f,0.f,0.f,0.f,0.f,0.f};
  v8f acc1 = acc0, acc2 = acc0, acc3 = acc0;

  const int arow = min(m0 + (lane & 15), Nrows - 1);    // clamp (stores are guarded)
  const float* aptr = X + (size_t)arow * K + half * 8;  // lanes>=16 start at K+8

  for (int k0 = 0; k0 < K; k0 += 32) {
    __syncthreads();
    // Stage W tile (32 K x 64 N) into LDS as bf16, transposed [n][k].
    // Two K-adjacent elements per thread -> v_cvt_pk_bf16_f32 + ds_store_b32.
    for (int i = threadIdx.x; i < 1024; i += 256) {
      int n = i & 63, k2 = (i >> 6) * 2;
      BF2U p;
      p.e[0] = (__bf16)W[(size_t)(k0 + k2) * M + n0 + n];
      p.e[1] = (__bf16)W[(size_t)(k0 + k2 + 1) * M + n0 + n];
      *(unsigned*)&Bs[n * 32 + k2] = p.u;
    }
    __syncthreads();

    // A fragment (16x32 bf16): lane holds K = {kb..kb+7, kb+16..kb+23}, kb = half*8.
    const float* ap = aptr + k0;
    if (k0 + 32 < K) __builtin_prefetch(ap + 32, 0, 1); // global_prefetch_b8
    float4 p0 = *(const float4*)(ap);
    float4 p1 = *(const float4*)(ap + 4);
    float4 p2 = *(const float4*)(ap + 16);
    float4 p3 = *(const float4*)(ap + 20);
    AFragU a;
    a.e[0]  = (__bf16)p0.x; a.e[1]  = (__bf16)p0.y; a.e[2]  = (__bf16)p0.z; a.e[3]  = (__bf16)p0.w;
    a.e[4]  = (__bf16)p1.x; a.e[5]  = (__bf16)p1.y; a.e[6]  = (__bf16)p1.z; a.e[7]  = (__bf16)p1.w;
    a.e[8]  = (__bf16)p2.x; a.e[9]  = (__bf16)p2.y; a.e[10] = (__bf16)p2.z; a.e[11] = (__bf16)p2.w;
    a.e[12] = (__bf16)p3.x; a.e[13] = (__bf16)p3.y; a.e[14] = (__bf16)p3.z; a.e[15] = (__bf16)p3.w;

    // B fragments (32x16 bf16): lane n holds K = {kb..kb+7, kb+16..kb+23}, kb = half*8.
    BFragU b0, b1, b2, b3;
    const int brow = (lane & 15) * 32 + half * 8;
    const uint4* bp0 = (const uint4*)&Bs[brow];
    const uint4* bp1 = (const uint4*)&Bs[brow + 16 * 32];
    const uint4* bp2 = (const uint4*)&Bs[brow + 32 * 32];
    const uint4* bp3 = (const uint4*)&Bs[brow + 48 * 32];
    b0.q[0] = bp0[0]; b0.q[1] = bp0[2];
    b1.q[0] = bp1[0]; b1.q[1] = bp1[2];
    b2.q[0] = bp2[0]; b2.q[1] = bp2[2];
    b3.q[0] = bp3[0]; b3.q[1] = bp3[2];

    acc0 = __builtin_amdgcn_wmma_f32_16x16x32_bf16(false, a.v, false, b0.v, (short)0, acc0, false, false);
    acc1 = __builtin_amdgcn_wmma_f32_16x16x32_bf16(false, a.v, false, b1.v, (short)0, acc1, false, false);
    acc2 = __builtin_amdgcn_wmma_f32_16x16x32_bf16(false, a.v, false, b2.v, (short)0, acc2, false, false);
    acc3 = __builtin_amdgcn_wmma_f32_16x16x32_bf16(false, a.v, false, b3.v, (short)0, acc3, false, false);
  }

  // C/D layout: VGPR r -> M = r + half*8, N = lane&15.
  const int rbase = m0 + half * 8;
  v8f accs[4] = {acc0, acc1, acc2, acc3};
#pragma unroll
  for (int t = 0; t < 4; ++t) {
    const int col = n0 + t * 16 + (lane & 15);
    const float bv = bias[col];
#pragma unroll
    for (int i = 0; i < 8; ++i) {
      int row = rbase + i;
      if (row < Nrows) Y[(size_t)row * M + col] = accs[t][i] + bv;
    }
  }
}

// ---------------------------------------------------------------------------
// Edge phase. One wave32 per edge, HC/32 channels per lane.
// ---------------------------------------------------------------------------
__device__ __forceinline__ unsigned f_orderable(float f) {
  unsigned u = __float_as_uint(f);
  return (u & 0x80000000u) ? ~u : (u | 0x80000000u);
}
__device__ __forceinline__ float f_unorderable(unsigned u) {
  return (u & 0x80000000u) ? __uint_as_float(u & 0x7FFFFFFFu) : __uint_as_float(~u);
}
__device__ __forceinline__ float lrelu_dot(float l, float r, float t) {
  float v = l + r;
  v = (v > 0.f) ? v : 0.2f * v;
  return v * t;
}

template <int H, int C>
__global__ void __launch_bounds__(256)
edge_score(const float* __restrict__ xl, const float* __restrict__ xr,
           const float* __restrict__ att, const int* __restrict__ src,
           const int* __restrict__ dst, float* __restrict__ score,
           unsigned* __restrict__ smax, int E) {
  const int lane = threadIdx.x & 31;
  const int e = blockIdx.x * 8 + (threadIdx.x >> 5);
  if (e >= E) return;
  constexpr int HC = H * C;
  constexpr int PER = HC / 32;           // channels per lane (all in one head)
  constexpr int G = 32 / H;              // lanes per head
  const size_t sb = (size_t)src[e] * HC;
  const size_t db = (size_t)dst[e] * HC;
  float s = 0.f;
  if constexpr (PER % 4 == 0) {
#pragma unroll
    for (int j = 0; j < PER; j += 4) {
      int c = lane * PER + j;
      float4 l = *(const float4*)&xl[sb + c];
      float4 r = *(const float4*)&xr[db + c];
      float4 t = *(const float4*)&att[c];
      s += lrelu_dot(l.x, r.x, t.x) + lrelu_dot(l.y, r.y, t.y) +
           lrelu_dot(l.z, r.z, t.z) + lrelu_dot(l.w, r.w, t.w);
    }
  } else {
#pragma unroll
    for (int j = 0; j < PER; ++j) {
      int c = lane * PER + j;
      s += lrelu_dot(xl[sb + c], xr[db + c], att[c]);
    }
  }
#pragma unroll
  for (int off = G >> 1; off > 0; off >>= 1) s += __shfl_xor(s, off);
  if ((lane & (G - 1)) == 0) {
    int h = lane / G;
    score[(size_t)e * H + h] = s;
    atomicMax(&smax[(size_t)dst[e] * H + h], f_orderable(s));
  }
}

template <int H>
__global__ void edge_softmax_den(float* __restrict__ score,
                                 const unsigned* __restrict__ smax,
                                 const int* __restrict__ dst,
                                 float* __restrict__ denom, int E) {
  int t = blockIdx.x * blockDim.x + threadIdx.x;
  if (t >= E * H) return;
  int e = t / H, h = t - e * H;
  float m = f_unorderable(smax[(size_t)dst[e] * H + h]);
  float ex = __expf(score[t] - m);
  score[t] = ex;                          // overwrite score with numerator
  atomicAdd(&denom[(size_t)dst[e] * H + h], ex);
}

template <int H, int C>
__global__ void __launch_bounds__(256)
edge_aggregate(const float* __restrict__ xl, const float* __restrict__ ex,
               const float* __restrict__ denom, const int* __restrict__ src,
               const int* __restrict__ dst, float* __restrict__ agg, int E) {
  const int lane = threadIdx.x & 31;
  const int e = blockIdx.x * 8 + (threadIdx.x >> 5);
  if (e >= E) return;
  constexpr int HC = H * C;
  constexpr int PER = HC / 32;
  const int d = dst[e];
  const size_t sb = (size_t)src[e] * HC;
  const int h = (lane * PER) / C;
  const float alpha = ex[(size_t)e * H + h] / (denom[(size_t)d * H + h] + 1e-16f);
  if constexpr (PER % 4 == 0) {
#pragma unroll
    for (int j = 0; j < PER; j += 4) {
      int c = lane * PER + j;
      float4 l = *(const float4*)&xl[sb + c];
      float* p = &agg[(size_t)d * HC + c];
      atomicAdd(p + 0, alpha * l.x);
      atomicAdd(p + 1, alpha * l.y);
      atomicAdd(p + 2, alpha * l.z);
      atomicAdd(p + 3, alpha * l.w);
    }
  } else {
#pragma unroll
    for (int j = 0; j < PER; ++j) {
      int c = lane * PER + j;
      atomicAdd(&agg[(size_t)d * HC + c], alpha * xl[sb + c]);
    }
  }
}

template <int HC>
__global__ void __launch_bounds__(256)
bias_layernorm(const float* __restrict__ agg, const float* __restrict__ bo,
               const float* __restrict__ g, const float* __restrict__ b,
               float* __restrict__ out, int N) {
  const int lane = threadIdx.x & 31;
  const int n = blockIdx.x * 8 + (threadIdx.x >> 5);
  if (n >= N) return;
  constexpr int PER = HC / 32;
  float v[PER];
  float s1 = 0.f, s2 = 0.f;
#pragma unroll
  for (int j = 0; j < PER; ++j) {
    int c = lane * PER + j;
    v[j] = agg[(size_t)n * HC + c] + bo[c];
    s1 += v[j];
    s2 += v[j] * v[j];
  }
#pragma unroll
  for (int off = 16; off > 0; off >>= 1) {
    s1 += __shfl_xor(s1, off);
    s2 += __shfl_xor(s2, off);
  }
  float mu = s1 / HC;
  float var = s2 / HC - mu * mu;
  float rs = rsqrtf(var + 1e-5f);
#pragma unroll
  for (int j = 0; j < PER; ++j) {
    int c = lane * PER + j;
    out[(size_t)n * HC + c] = (v[j] - mu) * rs * g[c] + b[c];
  }
}

__global__ void fill_u32(unsigned* __restrict__ p, unsigned v, unsigned long long n) {
  unsigned long long i = (unsigned long long)blockIdx.x * blockDim.x + threadIdx.x;
  if (i < n) p[i] = v;
}

// ---------------------------------------------------------------------------
extern "C" void kernel_launch(void* const* d_in, const int* in_sizes, int n_in,
                              void* d_out, int out_size, void* d_ws, size_t ws_size,
                              hipStream_t stream) {
  const int N = N_NODES, E = N_EDGES;
  const float* x   = (const float*)d_in[0];
  const int*   ei  = (const int*)d_in[1];
  const int*   src = ei;
  const int*   dstp = ei + E;
  const float* W1l = (const float*)d_in[3];  const float* b1l = (const float*)d_in[4];
  const float* W1r = (const float*)d_in[5];  const float* b1r = (const float*)d_in[6];
  const float* a1  = (const float*)d_in[7];  const float* bo1 = (const float*)d_in[8];
  const float* g1  = (const float*)d_in[9];  const float* be1 = (const float*)d_in[10];
  const float* W2l = (const float*)d_in[11]; const float* b2l = (const float*)d_in[12];
  const float* W2r = (const float*)d_in[13]; const float* b2r = (const float*)d_in[14];
  const float* a2  = (const float*)d_in[15]; const float* bo2 = (const float*)d_in[16];
  const float* g2  = (const float*)d_in[17]; const float* be2 = (const float*)d_in[18];
  const float* W3l = (const float*)d_in[19]; const float* b3l = (const float*)d_in[20];
  const float* W3r = (const float*)d_in[21]; const float* b3r = (const float*)d_in[22];
  const float* a3  = (const float*)d_in[23]; const float* bo3 = (const float*)d_in[24];
  const float* g3  = (const float*)d_in[25]; const float* be3 = (const float*)d_in[26];
  const float* linW = (const float*)d_in[27]; const float* linb = (const float*)d_in[28];

  // Workspace (floats): XL, XR, AG, H1, H2 each N*256; SC E*4; SM N*4 u32; DN N*4.
  float* ws = (float*)d_ws;
  const size_t NHC = (size_t)N * 256;
  float*    XL = ws;
  float*    XR = XL + NHC;
  float*    AG = XR + NHC;
  float*    H1 = AG + NHC;
  float*    H2 = H1 + NHC;
  float*    SC = H2 + NHC;
  unsigned* SM = (unsigned*)(SC + (size_t)E * 4);
  float*    DN = (float*)(SM + (size_t)N * 4);

  const dim3 blk(256);
  const int rowBlocks = (3125 + 7) / 8;        // 50000/16 = 3125 row tiles
  const int eb = (E + 7) / 8;                  // wave-per-edge blocks
  const int nb = (N + 7) / 8;                  // wave-per-node blocks

  auto gemm = [&](const float* Xp, const float* Wp, const float* bp, float* Yp,
                  int K, int M) {
    dim3 grid(rowBlocks, M / 64);
    wmma_gemm_bias<<<grid, blk, 0, stream>>>(Xp, Wp, bp, Yp, N, K, M);
  };
  auto fillu = [&](unsigned* p, unsigned v, unsigned long long n) {
    fill_u32<<<(unsigned)((n + 255) / 256), 256, 0, stream>>>(p, v, n);
  };

  const unsigned NEG_INF_ORD = 0x007FFFFFu;    // f_orderable(-inf)

  // -------- layer 1: GATv2 128 -> 4x64 --------
  gemm(x, W1l, b1l, XL, 128, 256);
  gemm(x, W1r, b1r, XR, 128, 256);
  fillu(SM, NEG_INF_ORD, (unsigned long long)N * 4);
  fillu((unsigned*)DN, 0u, (unsigned long long)N * 4);
  fillu((unsigned*)AG, 0u, (unsigned long long)NHC);
  edge_score<4, 64><<<eb, blk, 0, stream>>>(XL, XR, a1, src, dstp, SC, SM, E);
  edge_softmax_den<4><<<(E * 4 + 255) / 256, 256, 0, stream>>>(SC, SM, dstp, DN, E);
  edge_aggregate<4, 64><<<eb, blk, 0, stream>>>(XL, SC, DN, src, dstp, AG, E);
  bias_layernorm<256><<<nb, blk, 0, stream>>>(AG, bo1, g1, be1, H1, N);

  // -------- layer 2: GATv2 256 -> 4x64 --------
  gemm(H1, W2l, b2l, XL, 256, 256);
  gemm(H1, W2r, b2r, XR, 256, 256);
  fillu(SM, NEG_INF_ORD, (unsigned long long)N * 4);
  fillu((unsigned*)DN, 0u, (unsigned long long)N * 4);
  fillu((unsigned*)AG, 0u, (unsigned long long)NHC);
  edge_score<4, 64><<<eb, blk, 0, stream>>>(XL, XR, a2, src, dstp, SC, SM, E);
  edge_softmax_den<4><<<(E * 4 + 255) / 256, 256, 0, stream>>>(SC, SM, dstp, DN, E);
  edge_aggregate<4, 64><<<eb, blk, 0, stream>>>(XL, SC, DN, src, dstp, AG, E);
  bias_layernorm<256><<<nb, blk, 0, stream>>>(AG, bo2, g2, be2, H2, N);

  // -------- layer 3: GATv2 256 -> 1x64 --------
  gemm(H2, W3l, b3l, XL, 256, 64);
  gemm(H2, W3r, b3r, XR, 256, 64);
  fillu(SM, NEG_INF_ORD, (unsigned long long)N);
  fillu((unsigned*)DN, 0u, (unsigned long long)N);
  fillu((unsigned*)AG, 0u, (unsigned long long)N * 64);
  edge_score<1, 64><<<eb, blk, 0, stream>>>(XL, XR, a3, src, dstp, SC, SM, E);
  edge_softmax_den<1><<<(E + 255) / 256, 256, 0, stream>>>(SC, SM, dstp, DN, E);
  edge_aggregate<1, 64><<<eb, blk, 0, stream>>>(XL, SC, DN, src, dstp, AG, E);
  bias_layernorm<64><<<nb, blk, 0, stream>>>(AG, bo3, g3, be3, H1, N);

  // -------- final linear 64 -> 64 --------
  gemm(H1, linW, linb, (float*)d_out, 64, 64);
}
